// AdaptiveGraphPooling_48842368090161
// MI455X (gfx1250) — matrix operand order repaired and checked by reading, not compiled, not run on hardware
//
#include <hip/hip_runtime.h>

// ---------------------------------------------------------------------------
// AdaptiveGraphPooling on gfx1250 (MI455X)
//   phase 1: fused score MLP via v_wmma_f32_16x16x32_bf16 (bf16 A/B, f32 acc)
//   phase 2: exact deterministic radix top-k (k = N/2)
//   phase 3: pooled_x gather (float4, bandwidth bound)
//   phase 4: stable edge compaction (hierarchical scan + ballot prefix)
// ---------------------------------------------------------------------------

#define N_NODES 100000
#define C_IN    512
#define H_DIM   256
#define E_EDGES 3200000
#define DE_ATTR 16
#define TOPK    50000
#define MULT    1.0f

typedef __attribute__((ext_vector_type(16))) __bf16 v16bf;
typedef __attribute__((ext_vector_type(8)))  __bf16 v8bf;
typedef __attribute__((ext_vector_type(8)))  float  v8f;

#define KCHUNK    64
#define LDS_PITCH 72   // 64 + 8 pad (bf16 elems); row stride 144B, 16B aligned

// ============================ phase 1: score MLP ===========================
// Block = 256 threads = 8 waves. Block covers 128 rows (16 per wave).
// W1 chunk [256 x 64] staged in LDS as bf16. Each wave holds 16 accumulator
// tiles (16x16 f32 = 8 VGPR each) covering all H=256 outputs for its rows.
__global__ __launch_bounds__(256) void score_mlp_kernel(
    const float* __restrict__ x, const float* __restrict__ W1,
    const float* __restrict__ b1, const float* __restrict__ W2,
    const float* __restrict__ b2, float* __restrict__ scores)
{
  __shared__ __bf16 sW[H_DIM * LDS_PITCH];   // 36,864 bytes

  const int tid  = threadIdx.x;
  const int lane = tid & 31;
  const int wave = tid >> 5;
  const int l16  = lane & 15;
  const int half = lane >> 4;                // 0: K lo-block, 1: K hi-block
  const int rowbase = blockIdx.x * 128 + wave * 16;

  int arow = rowbase + l16;                  // A-matrix row for this lane
  if (arow >= N_NODES) arow = N_NODES - 1;   // clamp: keep EXEC all-ones
  const float* xrow = x + (size_t)arow * C_IN;

  // accumulators initialized with b1 broadcast (C layout: N = l16 per lane)
  v8f acc[16];
#pragma unroll
  for (int nh = 0; nh < 16; ++nh) {
    float bv = b1[nh * 16 + l16];
    v8f a = {bv, bv, bv, bv, bv, bv, bv, bv};
    acc[nh] = a;
  }

  for (int kc = 0; kc < C_IN / KCHUNK; ++kc) {
    const int kbase = kc * KCHUNK;

    // cooperative W1 chunk -> LDS (f32 -> bf16), 16 float4 per thread
    for (int i = 0; i < 16; ++i) {
      int v  = tid + i * 256;                // vec4 index 0..4095
      int h  = v >> 4;                       // 0..255
      int k4 = (v & 15) << 2;                // 0..60
      const float4 w = *(const float4*)(W1 + (size_t)h * C_IN + kbase + k4);
      __bf16* dst = &sW[h * LDS_PITCH + k4];
      dst[0] = (__bf16)w.x; dst[1] = (__bf16)w.y;
      dst[2] = (__bf16)w.z; dst[3] = (__bf16)w.w;
    }
    __syncthreads();

    if (kbase + KCHUNK < C_IN)
      __builtin_prefetch(xrow + kbase + KCHUNK, 0, 3);   // global_prefetch_b8

    // A fragments for this wave's rows (2 k-steps of 32)
    v16bf afrag[2];
#pragma unroll
    for (int ks = 0; ks < 2; ++ks) {
      const float* p0 = xrow + kbase + ks * 32 + (half ? 8 : 0);
      const float* p1 = xrow + kbase + ks * 32 + (half ? 24 : 16);
      float4 f0 = *(const float4*)(p0);
      float4 f1 = *(const float4*)(p0 + 4);
      float4 f2 = *(const float4*)(p1);
      float4 f3 = *(const float4*)(p1 + 4);
      v16bf a;
      a[0]=(__bf16)f0.x; a[1]=(__bf16)f0.y; a[2]=(__bf16)f0.z; a[3]=(__bf16)f0.w;
      a[4]=(__bf16)f1.x; a[5]=(__bf16)f1.y; a[6]=(__bf16)f1.z; a[7]=(__bf16)f1.w;
      a[8]=(__bf16)f2.x; a[9]=(__bf16)f2.y; a[10]=(__bf16)f2.z; a[11]=(__bf16)f2.w;
      a[12]=(__bf16)f3.x; a[13]=(__bf16)f3.y; a[14]=(__bf16)f3.z; a[15]=(__bf16)f3.w;
      afrag[ks] = a;
    }

#pragma unroll
    for (int nh = 0; nh < 16; ++nh) {
      const __bf16* wrow = &sW[(nh * 16 + l16) * LDS_PITCH];
#pragma unroll
      for (int ks = 0; ks < 2; ++ks) {
        const __bf16* q = wrow + ks * 32 + half * 8;     // 16B aligned
        v8bf blo = *(const v8bf*)(q);
        v8bf bhi = *(const v8bf*)(q + 16);
        v16bf bf = __builtin_shufflevector(blo, bhi,
                     0,1,2,3,4,5,6,7,8,9,10,11,12,13,14,15);
        acc[nh] = __builtin_amdgcn_wmma_f32_16x16x32_bf16(
            false, afrag[ks], false, bf, (short)0, acc[nh], false, false);
      }
    }
    __syncthreads();
  }

  // fused second linear: partial[m] = sum_n relu(h[m,n]) * W2[n]
  float partial[8];
#pragma unroll
  for (int m = 0; m < 8; ++m) partial[m] = 0.0f;
#pragma unroll
  for (int nh = 0; nh < 16; ++nh) {
    float w2v = W2[nh * 16 + l16];
#pragma unroll
    for (int m = 0; m < 8; ++m) {
      float h = acc[nh][m];
      h = h > 0.0f ? h : 0.0f;
      partial[m] += h * w2v;
    }
  }
  // reduce over the 16 lanes of each half-wave (rows 0..7 low / 8..15 high)
#pragma unroll
  for (int off = 8; off >= 1; off >>= 1) {
#pragma unroll
    for (int m = 0; m < 8; ++m)
      partial[m] += __shfl_xor(partial[m], off, 16);
  }
  if (l16 == 0) {
    const float bb = b2[0];
    const int mbase = rowbase + half * 8;
#pragma unroll
    for (int m = 0; m < 8; ++m) {
      int r = mbase + m;
      if (r < N_NODES) scores[r] = tanhf(partial[m] + bb);
    }
  }
}

// ====================== phase 2: exact radix top-k =========================
__device__ __forceinline__ unsigned int score_key(float s) {
  unsigned int u = __float_as_uint(s);
  return (u & 0x80000000u) ? ~u : (u | 0x80000000u);   // monotonic: big=large
}

__global__ void zero_kernel(unsigned int* p, int n) {
  int i = blockIdx.x * 256 + threadIdx.x;
  if (i < n) p[i] = 0u;
}

__global__ void hist_top11(const float* __restrict__ s, unsigned int* hist) {
  int i = blockIdx.x * 256 + threadIdx.x;
  if (i < N_NODES) atomicAdd(&hist[score_key(s[i]) >> 21], 1u);
}
__global__ void hist_mid11(const float* __restrict__ s,
                           const unsigned int* __restrict__ ctrl,
                           unsigned int* hist) {
  int i = blockIdx.x * 256 + threadIdx.x;
  if (i < N_NODES) {
    unsigned int k = score_key(s[i]);
    if ((k >> 21) == ctrl[0]) atomicAdd(&hist[(k >> 10) & 0x7FFu], 1u);
  }
}
__global__ void hist_low10(const float* __restrict__ s,
                           const unsigned int* __restrict__ ctrl,
                           unsigned int* hist) {
  int i = blockIdx.x * 256 + threadIdx.x;
  if (i < N_NODES) {
    unsigned int k = score_key(s[i]);
    if ((k >> 10) == ((ctrl[0] << 11) | ctrl[2]))
      atomicAdd(&hist[k & 0x3FFu], 1u);
  }
}

// walk bins from the top until cumulative count reaches target
__global__ void select_kernel(const unsigned int* __restrict__ hist,
                              unsigned int* ctrl, int nbins,
                              int tgt_slot, int bin_slot, int rem_slot,
                              int finalize) {
  if (threadIdx.x != 0 || blockIdx.x != 0) return;
  unsigned int target = (tgt_slot < 0) ? (unsigned int)TOPK : ctrl[tgt_slot];
  unsigned int cum = 0, rem = 0;
  int sel = 0;
  for (int b = nbins - 1; b >= 0; --b) {
    unsigned int c = hist[b];
    if (cum + c >= target) { sel = b; rem = target - cum; break; }
    cum += c;
  }
  ctrl[bin_slot] = (unsigned int)sel;
  ctrl[rem_slot] = rem;
  if (finalize)
    ctrl[4] = (ctrl[0] << 21) | (ctrl[2] << 10) | (unsigned int)sel;
}

// ================= hierarchical scans (nodes & edges) ======================
__global__ __launch_bounds__(1024) void node_count_kernel(
    const float* __restrict__ s, const unsigned int* __restrict__ ctrl,
    unsigned int* blk_gt, unsigned int* blk_tie) {
  __shared__ unsigned int sg[32], st[32];
  const unsigned int T = ctrl[4];
  int i = blockIdx.x * 1024 + threadIdx.x;
  int gt = 0, tie = 0;
  if (i < N_NODES) {
    unsigned int k = score_key(s[i]);
    gt = (k > T); tie = (k == T);
  }
  unsigned int mg = (unsigned int)__ballot(gt);
  unsigned int mt = (unsigned int)__ballot(tie);
  int lane = threadIdx.x & 31, wv = threadIdx.x >> 5;
  if (lane == 0) { sg[wv] = __popc(mg); st[wv] = __popc(mt); }
  __syncthreads();
  if (threadIdx.x == 0) {
    unsigned int tg = 0, tt = 0;
    for (int w = 0; w < 32; ++w) { tg += sg[w]; tt += st[w]; }
    blk_gt[blockIdx.x] = tg; blk_tie[blockIdx.x] = tt;
  }
}

// single-block exclusive scan over partials (n up to a few thousand)
__global__ __launch_bounds__(1024) void scan_partials_kernel(
    unsigned int* data, int n, unsigned int* total_out, int total_slot) {
  __shared__ unsigned int buf[1024];
  __shared__ unsigned int carry;
  if (threadIdx.x == 0) carry = 0;
  __syncthreads();
  for (int base = 0; base < n; base += 1024) {
    int i = base + threadIdx.x;
    unsigned int v = (i < n) ? data[i] : 0u;
    buf[threadIdx.x] = v;
    __syncthreads();
    for (int off = 1; off < 1024; off <<= 1) {
      unsigned int add = (threadIdx.x >= (unsigned)off) ? buf[threadIdx.x - off] : 0u;
      __syncthreads();
      buf[threadIdx.x] += add;
      __syncthreads();
    }
    unsigned int excl = buf[threadIdx.x] - v + carry;
    if (i < n) data[i] = excl;
    __syncthreads();
    if (threadIdx.x == 1023) carry += buf[1023];
    __syncthreads();
  }
  if (threadIdx.x == 0 && total_out) total_out[total_slot] = carry;
}

__global__ __launch_bounds__(1024) void node_emit_kernel(
    const float* __restrict__ s, const unsigned int* __restrict__ ctrl,
    const unsigned int* __restrict__ blk_gt,
    const unsigned int* __restrict__ blk_tie,
    int* node_map, unsigned int* perm_u32, float* perm_out) {
  __shared__ unsigned int sg[32], st[32];
  const unsigned int T = ctrl[4], k3 = ctrl[5];
  int i = blockIdx.x * 1024 + threadIdx.x;
  int gt = 0, tie = 0;
  if (i < N_NODES) {
    unsigned int k = score_key(s[i]);
    gt = (k > T); tie = (k == T);
  }
  unsigned int mg = (unsigned int)__ballot(gt);
  unsigned int mt = (unsigned int)__ballot(tie);
  int lane = threadIdx.x & 31, wv = threadIdx.x >> 5;
  unsigned int lm = (1u << lane) - 1u;
  unsigned int pg = __popc(mg & lm), pt = __popc(mt & lm);
  if (lane == 0) { sg[wv] = __popc(mg); st[wv] = __popc(mt); }
  __syncthreads();
  if (threadIdx.x == 0) {
    unsigned int ag = 0, at = 0;
    for (int w = 0; w < 32; ++w) {
      unsigned int g = sg[w], t = st[w];
      sg[w] = ag; st[w] = at; ag += g; at += t;
    }
  }
  __syncthreads();
  if (i < N_NODES) {
    unsigned int excl_gt  = blk_gt[blockIdx.x]  + sg[wv] + pg;
    unsigned int excl_tie = blk_tie[blockIdx.x] + st[wv] + pt;
    int selected = gt || (tie && excl_tie < k3);
    int new_id = -1;
    if (selected) {
      unsigned int tsel = excl_tie < k3 ? excl_tie : k3;
      new_id = (int)(excl_gt + tsel);
      perm_u32[new_id] = (unsigned int)i;
      perm_out[new_id] = (float)i;
    }
    node_map[i] = new_id;
  }
}

// ======================== phase 3: pooled_x gather =========================
__global__ void gather_kernel(const float* __restrict__ x,
                              const float* __restrict__ s,
                              const unsigned int* __restrict__ perm,
                              float* __restrict__ out) {
  long long idx = (long long)blockIdx.x * 256 + threadIdx.x;
  if (idx >= (long long)TOPK * (C_IN / 4)) return;
  int r  = (int)(idx / (C_IN / 4));
  int c4 = (int)(idx % (C_IN / 4)) * 4;
  unsigned int src = perm[r];
  float sc = s[src] * MULT;
  float4 v = *(const float4*)(x + (size_t)src * C_IN + c4);
  float4 o = {v.x * sc, v.y * sc, v.z * sc, v.w * sc};
  *(float4*)(out + (size_t)r * C_IN + c4) = o;
}

// ================== phase 4: stable edge compaction ========================
__global__ __launch_bounds__(1024) void edge_count_kernel(
    const long long* __restrict__ ei, const int* __restrict__ node_map,
    unsigned int* blk_cnt) {
  __shared__ unsigned int sc[32];
  long long e = (long long)blockIdx.x * 1024 + threadIdx.x;
  int valid = 0;
  if (e < E_EDGES) {
    long long a = ei[e], b = ei[(long long)E_EDGES + e];
    valid = (node_map[a] >= 0) && (node_map[b] >= 0);
  }
  unsigned int m = (unsigned int)__ballot(valid);
  int lane = threadIdx.x & 31, wv = threadIdx.x >> 5;
  if (lane == 0) sc[wv] = __popc(m);
  __syncthreads();
  if (threadIdx.x == 0) {
    unsigned int t = 0;
    for (int w = 0; w < 32; ++w) t += sc[w];
    blk_cnt[blockIdx.x] = t;
  }
}

__global__ __launch_bounds__(1024) void edge_emit_kernel(
    const long long* __restrict__ ei, const float* __restrict__ ea,
    const int* __restrict__ node_map, const unsigned int* __restrict__ blk_off,
    float* __restrict__ out_src, float* __restrict__ out_dst,
    float* __restrict__ out_attr) {
  __shared__ unsigned int sc[32];
  long long e = (long long)blockIdx.x * 1024 + threadIdx.x;
  int valid = 0, sn = -1, dn = -1;
  if (e < E_EDGES) {
    long long a = ei[e], b = ei[(long long)E_EDGES + e];
    sn = node_map[a]; dn = node_map[b];
    valid = (sn >= 0) && (dn >= 0);
  }
  unsigned int m = (unsigned int)__ballot(valid);
  int lane = threadIdx.x & 31, wv = threadIdx.x >> 5;
  unsigned int pg = __popc(m & ((1u << lane) - 1u));
  if (lane == 0) sc[wv] = __popc(m);
  __syncthreads();
  if (threadIdx.x == 0) {
    unsigned int a = 0;
    for (int w = 0; w < 32; ++w) { unsigned int g = sc[w]; sc[w] = a; a += g; }
  }
  __syncthreads();
  if (valid) {
    unsigned int pos = blk_off[blockIdx.x] + sc[wv] + pg;
    out_src[pos] = (float)sn;
    out_dst[pos] = (float)dn;
    const float4* ap = (const float4*)(ea + (size_t)e * DE_ATTR);
    float4* op = (float4*)(out_attr + (size_t)pos * DE_ATTR);
    op[0] = ap[0]; op[1] = ap[1]; op[2] = ap[2]; op[3] = ap[3];
  }
}

__global__ void edge_tail_kernel(const unsigned int* __restrict__ ctrl,
                                 float* __restrict__ out_src,
                                 float* __restrict__ out_dst,
                                 float* __restrict__ out_attr) {
  long long e = (long long)blockIdx.x * 256 + threadIdx.x;
  if (e >= E_EDGES) return;
  unsigned int nv = ctrl[6];
  if ((unsigned long long)e >= nv) {
    out_src[e] = -1.0f;
    out_dst[e] = -1.0f;
    float4 z = {0.f, 0.f, 0.f, 0.f};
    float4* op = (float4*)(out_attr + (size_t)e * DE_ATTR);
    op[0] = z; op[1] = z; op[2] = z; op[3] = z;
  }
}

// ================================= launch ==================================
extern "C" void kernel_launch(void* const* d_in, const int* in_sizes, int n_in,
                              void* d_out, int out_size, void* d_ws, size_t ws_size,
                              hipStream_t stream) {
  (void)in_sizes; (void)n_in; (void)out_size; (void)ws_size;
  const float*     x  = (const float*)d_in[0];
  const long long* ei = (const long long*)d_in[1];   // int64 edge_index
  const float*     ea = (const float*)d_in[2];
  const float*     W1 = (const float*)d_in[3];
  const float*     b1 = (const float*)d_in[4];
  const float*     W2 = (const float*)d_in[5];
  const float*     b2 = (const float*)d_in[6];

  // outputs, concatenated flat (indices/perm written as exact-int floats)
  float* out       = (float*)d_out;
  float* out_px    = out;                                   // 25,600,000
  float* out_src   = out_px  + (size_t)TOPK * C_IN;         //  3,200,000
  float* out_dst   = out_src + E_EDGES;                     //  3,200,000
  float* out_attr  = out_dst + E_EDGES;                     // 51,200,000
  float* out_perm  = out_attr + (size_t)E_EDGES * DE_ATTR;  //     50,000

  // workspace layout (4-byte units), ~1.04 MB total
  unsigned int* w = (unsigned int*)d_ws;
  float*        ws_scores = (float*)w;                 // [100000]
  int*          ws_map    = (int*)(w + 100000);        // [100000]
  unsigned int* ws_perm   = w + 200000;                // [50000]
  unsigned int* ws_hist1  = w + 250000;                // [2048]
  unsigned int* ws_hist2  = w + 252048;                // [2048]
  unsigned int* ws_hist3  = w + 254096;                // [1024]
  unsigned int* ws_ctrl   = w + 255120;                // [16]
  unsigned int* ws_nblk_g = w + 255136;                // [128]
  unsigned int* ws_nblk_t = w + 255264;                // [128]
  unsigned int* ws_eblk   = w + 255392;                // [3200]

  const int NB_NODE = (N_NODES + 1023) / 1024;   // 98
  const int NB_EDGE = (E_EDGES + 1023) / 1024;   // 3125

  // 0) zero hists + ctrl (5136 u32)
  zero_kernel<<<(5136 + 255) / 256, 256, 0, stream>>>(ws_hist1, 5136);

  // 1) fused score MLP (WMMA bf16)
  score_mlp_kernel<<<(N_NODES + 127) / 128, 256, 0, stream>>>(
      x, W1, b1, W2, b2, ws_scores);

  // 2) exact radix top-k: 11 + 11 + 10 bits
  const int HB = (N_NODES + 255) / 256;
  hist_top11<<<HB, 256, 0, stream>>>(ws_scores, ws_hist1);
  select_kernel<<<1, 32, 0, stream>>>(ws_hist1, ws_ctrl, 2048, -1, 0, 1, 0);
  hist_mid11<<<HB, 256, 0, stream>>>(ws_scores, ws_ctrl, ws_hist2);
  select_kernel<<<1, 32, 0, stream>>>(ws_hist2, ws_ctrl, 2048, 1, 2, 3, 0);
  hist_low10<<<HB, 256, 0, stream>>>(ws_scores, ws_ctrl, ws_hist3);
  select_kernel<<<1, 32, 0, stream>>>(ws_hist3, ws_ctrl, 1024, 3, 7, 5, 1);

  // 3) node selection scan -> perm + node_map
  node_count_kernel<<<NB_NODE, 1024, 0, stream>>>(ws_scores, ws_ctrl,
                                                  ws_nblk_g, ws_nblk_t);
  scan_partials_kernel<<<1, 1024, 0, stream>>>(ws_nblk_g, NB_NODE, nullptr, 0);
  scan_partials_kernel<<<1, 1024, 0, stream>>>(ws_nblk_t, NB_NODE, nullptr, 0);
  node_emit_kernel<<<NB_NODE, 1024, 0, stream>>>(ws_scores, ws_ctrl,
      ws_nblk_g, ws_nblk_t, ws_map, ws_perm, out_perm);

  // 4) pooled_x = x[perm] * scores[perm] * MULT
  gather_kernel<<<(TOPK * (C_IN / 4) + 255) / 256, 256, 0, stream>>>(
      x, ws_scores, ws_perm, out_px);

  // 5) stable edge compaction
  edge_count_kernel<<<NB_EDGE, 1024, 0, stream>>>(ei, ws_map, ws_eblk);
  scan_partials_kernel<<<1, 1024, 0, stream>>>(ws_eblk, NB_EDGE, ws_ctrl, 6);
  edge_emit_kernel<<<NB_EDGE, 1024, 0, stream>>>(ei, ea, ws_map, ws_eblk,
                                                 out_src, out_dst, out_attr);
  edge_tail_kernel<<<(E_EDGES + 255) / 256, 256, 0, stream>>>(
      ws_ctrl, out_src, out_dst, out_attr);
}
